// GATAE_26560077758611
// MI455X (gfx1250) — compile-verified, matrix operand order checked by reading, not run on hardware
//
#include <hip/hip_runtime.h>
#include <hip/hip_bf16.h>

// ---------------- problem constants (match reference) ----------------
#define NN_   12288
#define EE_   393216
#define IN_   256
#define HID_  128
#define NH_   4
#define OUT_  64
#define NEG_SLOPE_ 0.2f

typedef __bf16 v16bf __attribute__((ext_vector_type(16)));
typedef float  v8f   __attribute__((ext_vector_type(8)));

__device__ __forceinline__ unsigned short f2bf(float f) {
    union { float f; unsigned u; } x; x.f = f;
    unsigned r = x.u + 0x7FFFu + ((x.u >> 16) & 1u);   // round-to-nearest-even
    return (unsigned short)(r >> 16);
}

__device__ __forceinline__ int kpat(int e, int half) {
    // 16-bit A/B fragment K-order within a 32-wide K step (CDNA5 ISA 7.12.2)
    return (e < 8) ? (half * 8 + e) : (8 + half * 8 + e); // e>=8 -> 16 + 8*half + (e-8)
}

__device__ __forceinline__ void atomicMaxF(float* addr, float val) {
    unsigned* ua = (unsigned*)addr;
    unsigned old = *ua;
    while (true) {
        float f = __uint_as_float(old);
        if (f >= val) break;
        unsigned assumed = old;
        old = atomicCAS(ua, assumed, __float_as_uint(val));
        if (old == assumed) break;
    }
}

// ---------------- generic bf16 WMMA GEMM: C = act(A @ B + bias) ----------------
// A: [M,K] f32 row-major, B: [K,Nc] f32 row-major, C: [M,Nc] f32.
// ACT: 0=none, 1=ELU. K must be a multiple of 32, M a multiple of 128.
template <int ACT>
__global__ __launch_bounds__(256)
void gat_gemm_bf16(const float* __restrict__ A, const float* __restrict__ B,
                   const float* __restrict__ bias, float* __restrict__ C,
                   int M, int K, int Nc)
{
    constexpr int BM = 128, BN = 128, BK = 32;
    __shared__ unsigned short As[BM][BK + 2];
    __shared__ unsigned short Bs[BK][BN + 2];

    const int tid  = threadIdx.x;
    const int wave = tid >> 5;
    const int lane = tid & 31;
    const int half = lane >> 4;
    const int r    = lane & 15;
    const int rowBase = blockIdx.y * BM;
    const int colBase = blockIdx.x * BN;

    v8f acc[8] = {};

    for (int k0 = 0; k0 < K; k0 += BK) {
        // A tile 128x32
        for (int i = tid; i < BM * BK; i += 256) {
            int rr = i >> 5, cc = i & 31;
            int gr = rowBase + rr;
            float v = (gr < M) ? A[(size_t)gr * K + (k0 + cc)] : 0.0f;
            As[rr][cc] = f2bf(v);
        }
        // B tile 32x128 (zero-pad columns beyond Nc)
        for (int i = tid; i < BK * BN; i += 256) {
            int rr = i >> 7, cc = i & 127;
            int gc = colBase + cc;
            float v = (gc < Nc) ? B[(size_t)(k0 + rr) * Nc + gc] : 0.0f;
            Bs[rr][cc] = f2bf(v);
        }
        __syncthreads();

        union { unsigned short u[16]; v16bf v; } af;
#pragma unroll
        for (int e = 0; e < 16; e++) af.u[e] = As[wave * 16 + r][kpat(e, half)];

#pragma unroll
        for (int t = 0; t < 8; t++) {
            union { unsigned short u[16]; v16bf v; } bfr;
#pragma unroll
            for (int e = 0; e < 16; e++) bfr.u[e] = Bs[kpat(e, half)][t * 16 + r];
            acc[t] = __builtin_amdgcn_wmma_f32_16x16x32_bf16(
                false, af.v, false, bfr.v, (short)0, acc[t], false, false);
        }
        __syncthreads();
    }

    // epilogue: C/D layout -> VGPR i holds M = i + 8*half, N = lane&15
#pragma unroll
    for (int t = 0; t < 8; t++) {
        int gc = colBase + t * 16 + r;
        if (gc >= Nc) continue;
#pragma unroll
        for (int i = 0; i < 8; i++) {
            int gr = rowBase + wave * 16 + half * 8 + i;
            if (gr >= M) continue;
            float v = acc[t][i];
            if (bias) v += bias[gc];
            if (ACT == 1) {
                v = (v > 0.0f) ? v : (expf(v) - 1.0f);  // ELU
                __builtin_nontemporal_store(v, &C[(size_t)gr * Nc + gc]); // x_ output: streamed
            } else {
                C[(size_t)gr * Nc + gc] = v;            // intermediate: keep cacheable
            }
        }
    }
}

// ---------------- z @ z^T with sigmoid epilogue ----------------
// zbf: [Nn,64] bf16. out: [Nn,Nn] f32. Nn multiple of 128.
// Tile fills use the CDNA5 async global->LDS path (ASYNCcnt).
__global__ __launch_bounds__(256)
void gat_zzt_sigmoid(const unsigned short* __restrict__ zbf, float* __restrict__ out, int Nn)
{
    constexpr int BM = 128, K = 64;
    // no padding: rows are exactly 128 B so B128 async stores stay 16-byte aligned
    __shared__ __align__(16) unsigned short As[BM][K];   // row block of z
    __shared__ __align__(16) unsigned short Bs[BM][K];   // col block of z

    const int tid  = threadIdx.x;
    const int wave = tid >> 5;
    const int lane = tid & 31;
    const int half = lane >> 4;
    const int r    = lane & 15;
    const int rowBase = blockIdx.y * BM;
    const int colBase = blockIdx.x * BM;

    // async fill: 128 rows x 128B per tile, 16B chunks -> 1024 chunks/tile
    for (int i = tid; i < BM * 8; i += 256) {
        int rr = i >> 3, cc = i & 7;                      // cc indexes 16B chunks
        unsigned ldsA = (unsigned)(uintptr_t)(&As[rr][0]) + cc * 16;
        unsigned ldsB = (unsigned)(uintptr_t)(&Bs[rr][0]) + cc * 16;
        const unsigned short* ga = zbf + (size_t)(rowBase + rr) * K + cc * 8;
        const unsigned short* gb = zbf + (size_t)(colBase + rr) * K + cc * 8;
        asm volatile("global_load_async_to_lds_b128 %0, %1, off" :: "v"(ldsA), "v"(ga) : "memory");
        asm volatile("global_load_async_to_lds_b128 %0, %1, off" :: "v"(ldsB), "v"(gb) : "memory");
    }
    asm volatile("s_wait_asynccnt 0x0" ::: "memory");
    __syncthreads();

    v8f acc[8] = {};
#pragma unroll
    for (int ks = 0; ks < 2; ks++) {
        union { unsigned short u[16]; v16bf v; } af;
#pragma unroll
        for (int e = 0; e < 16; e++) af.u[e] = As[wave * 16 + r][ks * 32 + kpat(e, half)];
#pragma unroll
        for (int t = 0; t < 8; t++) {
            union { unsigned short u[16]; v16bf v; } bfr;
#pragma unroll
            for (int e = 0; e < 16; e++) bfr.u[e] = Bs[t * 16 + r][ks * 32 + kpat(e, half)];
            acc[t] = __builtin_amdgcn_wmma_f32_16x16x32_bf16(
                false, af.v, false, bfr.v, (short)0, acc[t], false, false);
        }
    }

    // streaming epilogue: 604 MB output, write-once -> non-temporal stores
#pragma unroll
    for (int t = 0; t < 8; t++) {
        int gc = colBase + t * 16 + r;
#pragma unroll
        for (int i = 0; i < 8; i++) {
            int gr = rowBase + wave * 16 + half * 8 + i;
            float v = acc[t][i];
            v = 1.0f / (1.0f + expf(-v));       // sigmoid
            __builtin_nontemporal_store(v, &out[(size_t)gr * Nn + gc]);
        }
    }
}

// ---------------- small helper kernels ----------------
__global__ void gat_fill(float* p, int n, float v) {
    int i = blockIdx.x * blockDim.x + threadIdx.x;
    if (i < n) p[i] = v;
}

// out[n*C + c] = bias[c]   (aggregation accumulator init)
__global__ void gat_init_bias(float* out, const float* bias, int total, int C) {
    int i = blockIdx.x * blockDim.x + threadIdx.x;
    if (i < total) out[i] = bias[i % C];
}

// a_src[n*H+h] = dot(h_lin[n,h,:], att_src[h,:]); same for dst
__global__ void gat_attn_coef(const float* __restrict__ hlin,
                              const float* __restrict__ att_s,
                              const float* __restrict__ att_d,
                              float* __restrict__ a_s, float* __restrict__ a_d,
                              int Nn, int H, int C) {
    int i = blockIdx.x * blockDim.x + threadIdx.x;
    if (i >= Nn * H) return;
    int n = i / H, h = i % H;
    const float* hp = hlin + (size_t)n * H * C + (size_t)h * C;
    float ss = 0.0f, sd = 0.0f;
    for (int c = 0; c < C; c++) { float v = hp[c]; ss += v * att_s[h * C + c]; sd += v * att_d[h * C + c]; }
    a_s[i] = ss; a_d[i] = sd;
}

// pass1: leaky-relu logits + segment max into m[d,h]
__global__ void gat_edge_pass1(const int* __restrict__ ei,
                               const float* __restrict__ a_s, const float* __restrict__ a_d,
                               float* __restrict__ eraw, float* __restrict__ m,
                               int E, int Etot, int H) {
    int e = blockIdx.x * blockDim.x + threadIdx.x;
    if (e >= Etot) return;
    int s = (e < E) ? ei[e] : (e - E);
    int d = (e < E) ? ei[E + e] : (e - E);
    for (int h = 0; h < H; h++) {
        float v = a_s[s * H + h] + a_d[d * H + h];
        v = (v > 0.0f) ? v : NEG_SLOPE_ * v;
        eraw[(size_t)e * H + h] = v;
        atomicMaxF(&m[d * H + h], v);
    }
}

// pass2: exp(e - m[d]) and segment sum into denom[d,h]
__global__ void gat_edge_pass2(const int* __restrict__ ei,
                               float* __restrict__ eraw, const float* __restrict__ m,
                               float* __restrict__ denom,
                               int E, int Etot, int H) {
    int e = blockIdx.x * blockDim.x + threadIdx.x;
    if (e >= Etot) return;
    int d = (e < E) ? ei[E + e] : (e - E);
    for (int h = 0; h < H; h++) {
        float x = expf(eraw[(size_t)e * H + h] - m[d * H + h]);
        eraw[(size_t)e * H + h] = x;
        atomicAdd(&denom[d * H + h], x);
    }
}

// pass3: out[d,h,:] += h_lin[s,h,:] * alpha
__global__ void gat_edge_pass3(const int* __restrict__ ei,
                               const float* __restrict__ hlin,
                               const float* __restrict__ eexp, const float* __restrict__ denom,
                               float* __restrict__ out,
                               int E, int Etot, int H, int C) {
    int i = blockIdx.x * blockDim.x + threadIdx.x;
    if (i >= Etot * H) return;
    int e = i / H, h = i % H;
    int s = (e < E) ? ei[e] : (e - E);
    int d = (e < E) ? ei[E + e] : (e - E);
    float w = eexp[(size_t)e * H + h] / (denom[d * H + h] + 1e-16f);
    const float* hp = hlin + (size_t)s * H * C + (size_t)h * C;
    float* op = out + (size_t)d * H * C + (size_t)h * C;
    for (int c = 0; c < C; c++) atomicAdd(&op[c], hp[c] * w);
}

// row-normalize h2 -> z (f32 into d_out slice) + bf16 copy for WMMA
__global__ void gat_normalize(const float* __restrict__ h2,
                              float* __restrict__ z_out, unsigned short* __restrict__ zbf,
                              int Nn) {
    int n = blockIdx.x * blockDim.x + threadIdx.x;
    if (n >= Nn) return;
    float s = 0.0f;
    for (int c = 0; c < OUT_; c++) { float v = h2[(size_t)n * OUT_ + c]; s += v * v; }
    float inv = 1.0f / fmaxf(sqrtf(s), 1e-12f);
    for (int c = 0; c < OUT_; c++) {
        float v = h2[(size_t)n * OUT_ + c] * inv;
        z_out[(size_t)n * OUT_ + c] = v;
        zbf[(size_t)n * OUT_ + c] = f2bf(v);
    }
}

// ---------------- host-side orchestration ----------------
extern "C" void kernel_launch(void* const* d_in, const int* in_sizes, int n_in,
                              void* d_out, int out_size, void* d_ws, size_t ws_size,
                              hipStream_t stream) {
    const float* x        = (const float*)d_in[0];
    const int*   ei       = (const int*)  d_in[1];
    const float* W1       = (const float*)d_in[2];
    const float* att_src1 = (const float*)d_in[3];
    const float* att_dst1 = (const float*)d_in[4];
    const float* b1       = (const float*)d_in[5];
    const float* W2       = (const float*)d_in[6];
    const float* att_src2 = (const float*)d_in[7];
    const float* att_dst2 = (const float*)d_in[8];
    const float* b2       = (const float*)d_in[9];
    const float* Wl       = (const float*)d_in[10];
    const float* bl       = (const float*)d_in[11];

    const int N = NN_, E = EE_, Etot = EE_ + NN_;
    const int HC1 = NH_ * HID_;            // 512

    // d_out layout: A_pred [N*N] | z [N*OUT] | x_ [N*IN]
    float* A_pred = (float*)d_out;
    float* z_out  = A_pred + (size_t)N * N;
    float* x_out  = z_out + (size_t)N * OUT_;

    // workspace layout (floats)
    float* ws = (float*)d_ws;
    float* h1_lin  = ws;                 ws += (size_t)N * HC1;      // 6.29M
    float* h1_out  = ws;                 ws += (size_t)N * HC1;      // 6.29M
    float* a_s1    = ws;                 ws += (size_t)N * NH_;
    float* a_d1    = ws;                 ws += (size_t)N * NH_;
    float* m1      = ws;                 ws += (size_t)N * NH_;
    float* dn1     = ws;                 ws += (size_t)N * NH_;
    float* eraw1   = ws;                 ws += (size_t)Etot * NH_;   // 1.62M
    float* h2_lin  = ws;                 ws += (size_t)N * OUT_;
    float* h2_out  = ws;                 ws += (size_t)N * OUT_;
    float* a_s2    = ws;                 ws += (size_t)N;
    float* a_d2    = ws;                 ws += (size_t)N;
    float* m2      = ws;                 ws += (size_t)N;
    float* dn2     = ws;                 ws += (size_t)N;
    float* eraw2   = ws;                 ws += (size_t)Etot;
    unsigned short* zbf = (unsigned short*)ws;

    const int TB = 256;
    auto blocks = [](int n, int tb) { return (n + tb - 1) / tb; };

    // ---- layer 1 linear: h1_lin = x @ W1 ----
    gat_gemm_bf16<0><<<dim3(HC1 / 128, N / 128), TB, 0, stream>>>(x, W1, nullptr, h1_lin, N, IN_, HC1);

    // ---- layer 1 attention ----
    gat_attn_coef<<<blocks(N * NH_, TB), TB, 0, stream>>>(h1_lin, att_src1, att_dst1, a_s1, a_d1, N, NH_, HID_);
    gat_fill<<<blocks(N * NH_, TB), TB, 0, stream>>>(m1, N * NH_, -1e30f);
    gat_fill<<<blocks(N * NH_, TB), TB, 0, stream>>>(dn1, N * NH_, 0.0f);
    gat_init_bias<<<blocks(N * HC1, TB), TB, 0, stream>>>(h1_out, b1, N * HC1, HC1);
    gat_edge_pass1<<<blocks(Etot, TB), TB, 0, stream>>>(ei, a_s1, a_d1, eraw1, m1, E, Etot, NH_);
    gat_edge_pass2<<<blocks(Etot, TB), TB, 0, stream>>>(ei, eraw1, m1, dn1, E, Etot, NH_);
    gat_edge_pass3<<<blocks(Etot * NH_, TB), TB, 0, stream>>>(ei, h1_lin, eraw1, dn1, h1_out, E, Etot, NH_, HID_);

    // ---- layer 2 linear: h2_lin = h1_out @ W2 ----
    gat_gemm_bf16<0><<<dim3(1, N / 128), TB, 0, stream>>>(h1_out, W2, nullptr, h2_lin, N, HC1, OUT_);

    // ---- layer 2 attention (H=1, C=64, concat=False mean over 1 head == identity) ----
    gat_attn_coef<<<blocks(N, TB), TB, 0, stream>>>(h2_lin, att_src2, att_dst2, a_s2, a_d2, N, 1, OUT_);
    gat_fill<<<blocks(N, TB), TB, 0, stream>>>(m2, N, -1e30f);
    gat_fill<<<blocks(N, TB), TB, 0, stream>>>(dn2, N, 0.0f);
    gat_init_bias<<<blocks(N * OUT_, TB), TB, 0, stream>>>(h2_out, b2, N * OUT_, OUT_);
    gat_edge_pass1<<<blocks(Etot, TB), TB, 0, stream>>>(ei, a_s2, a_d2, eraw2, m2, E, Etot, 1);
    gat_edge_pass2<<<blocks(Etot, TB), TB, 0, stream>>>(ei, eraw2, m2, dn2, E, Etot, 1);
    gat_edge_pass3<<<blocks(Etot, TB), TB, 0, stream>>>(ei, h2_lin, eraw2, dn2, h2_out, E, Etot, 1, OUT_);

    // ---- normalize -> z (writes the z slice of d_out) + bf16 copy ----
    gat_normalize<<<blocks(N, TB), TB, 0, stream>>>(h2_out, z_out, zbf, N);

    // ---- x_ = elu(z @ Wl + bl) ----
    gat_gemm_bf16<1><<<dim3(IN_ / 128, N / 128), TB, 0, stream>>>(z_out, Wl, bl, x_out, N, OUT_, IN_);

    // ---- A_pred = sigmoid(z @ z^T)  (bandwidth-bound: 604 MB out) ----
    gat_zzt_sigmoid<<<dim3(N / 128, N / 128), TB, 0, stream>>>(zbf, A_pred, N);
}
